// Bahdanau_Attention_1589137900140
// MI455X (gfx1250) — compile-verified, matrix operand order checked by reading, not run on hardware
//
#include <hip/hip_runtime.h>
#include <hip/hip_bf16.h>
#include <math.h>

#define BATCH 32
#define SEQ   2048
#define KDIM  1024
#define ADIM  1024

#define MTILE   64    // seq rows per workgroup
#define NCHUNK  128   // attention-dim columns per chunk (4 tiles per wave)
#define KSLICE  64    // K elements staged in LDS per slice
#define NSLICES (KDIM / KSLICE)
#define LPITCH  72    // LDS row pitch (bf16 elems), 144B rows, 16B-aligned

#define SSPLIT 8
#define SCHUNK (SEQ / SSPLIT)

typedef __attribute__((ext_vector_type(16))) __bf16 v16bf;
typedef __attribute__((ext_vector_type(8)))  float  v8f;
typedef __attribute__((ext_vector_type(4)))  int    v4i;

#if __has_builtin(__builtin_amdgcn_global_load_async_to_lds_b128) && \
    __has_builtin(__builtin_amdgcn_s_wait_asynccnt)
#define ASYNC_WE 1
#else
#define ASYNC_WE 0
#endif

union FragBF16 {
  v16bf v;
  int4  q[2];
};

// fp32 -> bf16 round-to-nearest-even, packed pair into one dword
__device__ __forceinline__ unsigned int pack_bf16(float lo, float hi) {
  union { float f; unsigned int u; } a, b;
  a.f = lo; b.f = hi;
  unsigned int rl = (a.u + 0x7FFFu + ((a.u >> 16) & 1u)) >> 16;
  unsigned int rh = (b.u + 0x7FFFu + ((b.u >> 16) & 1u)) >> 16;
  return (rl & 0xFFFFu) | (rh << 16);
}

// Single-instruction tanh on gfx1250 (V_TANH_F32); branch-free fallback.
#if __has_builtin(__builtin_amdgcn_tanhf)
__device__ __forceinline__ float fast_tanh(float x) {
  return __builtin_amdgcn_tanhf(x);
}
#else
__device__ __forceinline__ float fast_tanh(float x) {
  float ax = fabsf(x);
  float t  = __expf(-2.0f * ax);
  float r  = (1.0f - t) * __builtin_amdgcn_rcpf(1.0f + t);
  return copysignf(r, x);
}
#endif

// ---------------------------------------------------------------------------
// Kernel 0: one-time W_encoder fp32 -> bf16 (2 MB, L2-resident thereafter)
// ---------------------------------------------------------------------------
__global__ void __launch_bounds__(256)
convert_we_kernel(const float* __restrict__ We,
                  unsigned short* __restrict__ WeBf) {
  size_t i = (size_t)blockIdx.x * 256 + threadIdx.x;  // float4 index
  float4 vv = ((const float4*)We)[i];
  uint2 p;
  p.x = pack_bf16(vv.x, vv.y);
  p.y = pack_bf16(vv.z, vv.w);
  ((uint2*)WeBf)[i] = p;
}

// ---------------------------------------------------------------------------
// Kernel 1: V[b][a] = sum_q q[b][qd] * W_decoder[a][qd]   (tiny, fp32 VALU)
// ---------------------------------------------------------------------------
__global__ void __launch_bounds__(256)
vdec_kernel(const float* __restrict__ q, const float* __restrict__ Wd,
            float* __restrict__ Vbuf) {
  const int b = blockIdx.y;
  const int a = blockIdx.x * 256 + threadIdx.x;
  const float4* qp = (const float4*)(q + (size_t)b * KDIM);
  const float4* wp = (const float4*)(Wd + (size_t)a * KDIM);
  float acc = 0.f;
#pragma unroll 4
  for (int i = 0; i < KDIM / 4; ++i) {
    float4 qa = qp[i], wa = wp[i];
    acc += qa.x * wa.x + qa.y * wa.y + qa.z * wa.z + qa.w * wa.w;
  }
  Vbuf[(size_t)b * ADIM + a] = acc;
}

// ---------------------------------------------------------------------------
// Staging helpers for the score kernel pipeline
// ---------------------------------------------------------------------------
__device__ __forceinline__ void load_key_regs(float4 kreg[4],
                                              const float* __restrict__ keyBase,
                                              int ks, int tid) {
#pragma unroll
  for (int i = 0; i < 4; ++i) {
    int fi = tid + i * 256;            // 64 rows x 16 float4
    int row = fi >> 4, c4 = fi & 15;
    kreg[i] = *(const float4*)(keyBase + (size_t)row * KDIM + ks * KSLICE +
                               c4 * 4);
  }
}

__device__ __forceinline__ void store_key_lds(unsigned short (*kl)[LPITCH],
                                              const float4 kreg[4], int tid) {
#pragma unroll
  for (int i = 0; i < 4; ++i) {
    int fi = tid + i * 256;
    int row = fi >> 4, c4 = fi & 15;
    unsigned int* kd = (unsigned int*)&kl[row][c4 * 4];
    kd[0] = pack_bf16(kreg[i].x, kreg[i].y);
    kd[1] = pack_bf16(kreg[i].z, kreg[i].w);
  }
}

#if ASYNC_WE
// DMA the bf16 W_encoder slice straight into LDS (ASYNCcnt-tracked).
__device__ __forceinline__ void issue_we_async(
    unsigned short (*wl)[LPITCH], const unsigned short* __restrict__ WeBf,
    int nc, int ks, int tid) {
#pragma unroll
  for (int i = 0; i < 4; ++i) {
    int fi = tid + i * 256;            // 128 rows x 8 int4
    int row = fi >> 3, c8 = fi & 7;
    const unsigned short* g = WeBf + (size_t)(nc * NCHUNK + row) * KDIM +
                              ks * KSLICE + c8 * 8;
    __builtin_amdgcn_global_load_async_to_lds_b128(
        (v4i*)g, (v4i*)&wl[row][c8 * 8], 0, 0);
  }
}
#else
__device__ __forceinline__ void load_we_regs(int4 wreg[4],
                                             const unsigned short* __restrict__
                                                 WeBf,
                                             int nc, int ks, int tid) {
#pragma unroll
  for (int i = 0; i < 4; ++i) {
    int fi = tid + i * 256;
    int row = fi >> 3, c8 = fi & 7;
    wreg[i] = *(const int4*)(WeBf + (size_t)(nc * NCHUNK + row) * KDIM +
                             ks * KSLICE + c8 * 8);
  }
}
__device__ __forceinline__ void store_we_lds(unsigned short (*wl)[LPITCH],
                                             const int4 wreg[4], int tid) {
#pragma unroll
  for (int i = 0; i < 4; ++i) {
    int fi = tid + i * 256;
    int row = fi >> 3, c8 = fi & 7;
    *(int4*)&wl[row][c8 * 8] = wreg[i];
  }
}
#endif

// ---------------------------------------------------------------------------
// Kernel 2: fused  score[b][s] = v . tanh(key[b][s]·We^T + V[b])
// Double-buffered software pipeline: WMMAs on slice k overlap the key
// global->reg loads and the async-to-LDS DMA of slice k+1.
// ---------------------------------------------------------------------------
__global__ void __launch_bounds__(256)
score_kernel(const float* __restrict__ key,
             const unsigned short* __restrict__ WeBf,
             const float* __restrict__ Vbuf, const float* __restrict__ vvec,
             float* __restrict__ scores) {
  __shared__ alignas(16) unsigned short keyLds[2][MTILE][LPITCH];   // 18 KB
  __shared__ alignas(16) unsigned short weLds[2][NCHUNK][LPITCH];   // 36 KB
  __shared__ float VLds[ADIM];                                     // 4 KB
  __shared__ float vLds[ADIM];                                     // 4 KB
  __shared__ float scoreLds[MTILE];

  const int tid  = threadIdx.x;
  const int b    = blockIdx.y;
  const int s0   = blockIdx.x * MTILE;
  const int wave = tid >> 5;
  const int lane = tid & 31;
  const int lh   = lane >> 4;          // half-wave select
  const int ln   = lane & 15;
  const int m0   = (wave & 3) * 16;    // this wave's M tile (seq rows)
  const int ng64 = (wave >> 2) * 64;   // this wave's N group (4 tiles)

  for (int i = tid; i < ADIM; i += 256) {
    VLds[i] = Vbuf[(size_t)b * ADIM + i];
    vLds[i] = vvec[i];
  }
  if (tid < MTILE) scoreLds[tid] = 0.0f;

  const float* keyBase = key + ((size_t)(b * SEQ + s0)) * KDIM;

  float sacc[8];
#pragma unroll
  for (int r = 0; r < 8; ++r) sacc[r] = 0.0f;

  for (int nc = 0; nc < ADIM / NCHUNK; ++nc) {
    v8f cacc[4];
#pragma unroll
    for (int t = 0; t < 4; ++t) cacc[t] = 0;

    // pipeline prologue: stage slice 0
    float4 kreg[4];
    load_key_regs(kreg, keyBase, 0, tid);
#if ASYNC_WE
    issue_we_async(weLds[0], WeBf, nc, 0, tid);
#else
    int4 wreg[4];
    load_we_regs(wreg, WeBf, nc, 0, tid);
#endif

    for (int ks = 0; ks < NSLICES; ++ks) {
      const int cur = ks & 1;
      store_key_lds(keyLds[cur], kreg, tid);
#if ASYNC_WE
      __builtin_amdgcn_s_wait_asynccnt(0);   // our DMA of slice ks landed
#else
      store_we_lds(weLds[cur], wreg, tid);
#endif
      __syncthreads();                       // slice ks visible to all waves
      if (ks + 1 < NSLICES) {                // prefetch slice ks+1
        load_key_regs(kreg, keyBase, ks + 1, tid);
#if ASYNC_WE
        issue_we_async(weLds[cur ^ 1], WeBf, nc, ks + 1, tid);
#else
        load_we_regs(wreg, WeBf, nc, ks + 1, tid);
#endif
      }
#pragma unroll
      for (int kk = 0; kk < KSLICE; kk += 32) {
        // 16-bit fragment layout (ISA 7.12.2): lanes 0-15 K{0-7,16-23},
        // lanes 16-31 K{8-15,24-31}; two aligned 16B LDS loads each.
        const int baseK = kk + (lh ? 8 : 0);
        FragBF16 a;
        const unsigned short* ap = &keyLds[cur][m0 + ln][baseK];
        a.q[0] = *(const int4*)ap;
        a.q[1] = *(const int4*)(ap + 16);
#pragma unroll
        for (int t = 0; t < 4; ++t) {
          FragBF16 bb;
          const unsigned short* bp = &weLds[cur][ng64 + t * 16 + ln][baseK];
          bb.q[0] = *(const int4*)bp;
          bb.q[1] = *(const int4*)(bp + 16);
          cacc[t] = __builtin_amdgcn_wmma_f32_16x16x32_bf16(
              false, a.v, false, bb.v, (short)0, cacc[t], false, false);
        }
      }
    }
    // epilogue: score += v[n] * tanh(U + V[n]) for this 128-wide N chunk.
    // C layout: element (M = r + 8*lh, N = ln) in c[r].
    const int nbase = nc * NCHUNK + ng64 + ln;
#pragma unroll
    for (int t = 0; t < 4; ++t) {
      const int n = nbase + t * 16;
      const float Vn = VLds[n], vn = vLds[n];
#pragma unroll
      for (int r = 0; r < 8; ++r) {
        sacc[r] += fast_tanh(cacc[t][r] + Vn) * vn;
      }
    }
  }

  // reduce over the 16 N-lanes of each half-wave, then combine waves in LDS
#pragma unroll
  for (int r = 0; r < 8; ++r) {
    float x = sacc[r];
    x += __shfl_xor(x, 1, 16);
    x += __shfl_xor(x, 2, 16);
    x += __shfl_xor(x, 4, 16);
    x += __shfl_xor(x, 8, 16);
    if (ln == 0) atomicAdd(&scoreLds[m0 + lh * 8 + r], x);  // ds_add_f32
  }
  __syncthreads();
  if (tid < MTILE) scores[(size_t)b * SEQ + s0 + tid] = scoreLds[tid];
}

// ---------------------------------------------------------------------------
// Kernel 3: in-place softmax over S per batch (one workgroup per batch)
// ---------------------------------------------------------------------------
__global__ void __launch_bounds__(256)
softmax_kernel(float* __restrict__ scores) {
  __shared__ float red[256];
  const int b = blockIdx.x, tid = threadIdx.x;
  float* s = scores + (size_t)b * SEQ;

  float lmax = -1e30f;
  for (int i = tid; i < SEQ; i += 256) lmax = fmaxf(lmax, s[i]);
  red[tid] = lmax;
  __syncthreads();
  for (int off = 128; off > 0; off >>= 1) {
    if (tid < off) red[tid] = fmaxf(red[tid], red[tid + off]);
    __syncthreads();
  }
  const float m = red[0];
  __syncthreads();

  float lsum = 0.f;
  for (int i = tid; i < SEQ; i += 256) {
    float e = __expf(s[i] - m);
    s[i] = e;
    lsum += e;
  }
  red[tid] = lsum;
  __syncthreads();
  for (int off = 128; off > 0; off >>= 1) {
    if (tid < off) red[tid] += red[tid + off];
    __syncthreads();
  }
  const float inv = 1.0f / red[0];
  for (int i = tid; i < SEQ; i += 256) s[i] *= inv;
}

// ---------------------------------------------------------------------------
// Kernel 4a: partial context over an S-chunk (deterministic, no atomics)
// ---------------------------------------------------------------------------
__global__ void __launch_bounds__(256)
context_partial_kernel(const float* __restrict__ key,
                       const float* __restrict__ w,
                       float* __restrict__ partial) {
  __shared__ float wsh[SCHUNK];
  const int b = blockIdx.y;
  const int z = blockIdx.z;
  const int k = blockIdx.x * 256 + threadIdx.x;
  const int sBase = z * SCHUNK;
  for (int i = threadIdx.x; i < SCHUNK; i += 256)
    wsh[i] = w[(size_t)b * SEQ + sBase + i];
  __syncthreads();
  const float* kp = key + ((size_t)b * SEQ + sBase) * KDIM + k;
  float acc = 0.f;
#pragma unroll 8
  for (int s = 0; s < SCHUNK; ++s) acc += wsh[s] * kp[(size_t)s * KDIM];
  partial[((size_t)z * BATCH + b) * KDIM + k] = acc;
}

// ---------------------------------------------------------------------------
// Kernel 4b: reduce the S-split partials (fixed order -> deterministic)
// ---------------------------------------------------------------------------
__global__ void __launch_bounds__(256)
context_reduce_kernel(const float* __restrict__ partial,
                      float* __restrict__ out) {
  const int b = blockIdx.y;
  const int k = blockIdx.x * 256 + threadIdx.x;
  float acc = 0.f;
#pragma unroll
  for (int z = 0; z < SSPLIT; ++z)
    acc += partial[((size_t)z * BATCH + b) * KDIM + k];
  out[(size_t)b * KDIM + k] = acc;
}

// ---------------------------------------------------------------------------
extern "C" void kernel_launch(void* const* d_in, const int* in_sizes, int n_in,
                              void* d_out, int out_size, void* d_ws,
                              size_t ws_size, hipStream_t stream) {
  const float* key = (const float*)d_in[0];
  const float* q   = (const float*)d_in[1];
  const float* We  = (const float*)d_in[2];
  const float* Wd  = (const float*)d_in[3];
  const float* v   = (const float*)d_in[4];
  float* out = (float*)d_out;

  unsigned short* WeBf = (unsigned short*)d_ws;          // [A][K] bf16 = 2 MB
  float* Vbuf    = (float*)(WeBf + (size_t)ADIM * KDIM); // [B][A]   = 128 KB
  float* scores  = Vbuf + BATCH * ADIM;                  // [B][S]   = 256 KB
  float* partial = scores + BATCH * SEQ;                 // [8][B][K] = 1 MB

  convert_we_kernel<<<(ADIM * KDIM / 4) / 256, 256, 0, stream>>>(We, WeBf);
  vdec_kernel<<<dim3(ADIM / 256, BATCH), 256, 0, stream>>>(q, Wd, Vbuf);
  score_kernel<<<dim3(SEQ / MTILE, BATCH), 256, 0, stream>>>(key, WeBf, Vbuf,
                                                             v, scores);
  softmax_kernel<<<BATCH, 256, 0, stream>>>(scores);
  context_partial_kernel<<<dim3(KDIM / 256, BATCH, SSPLIT), 256, 0, stream>>>(
      key, scores, partial);
  context_reduce_kernel<<<dim3(KDIM / 256, BATCH), 256, 0, stream>>>(partial,
                                                                     out);
}
